// Attention_16638703305084
// MI455X (gfx1250) — matrix operand order repaired and checked by reading, not compiled, not run on hardware
//
#include <hip/hip_runtime.h>
#include <hip/hip_bf16.h>
#include <math.h>

typedef __attribute__((ext_vector_type(16))) __bf16 v16bf;
typedef __attribute__((ext_vector_type(8)))  float  v8f;
typedef __attribute__((ext_vector_type(2)))  float  v2f;
typedef __attribute__((ext_vector_type(4)))  int    v4i;

typedef __attribute__((address_space(1))) v4i* gptr4;   // global 128-bit chunk
typedef __attribute__((address_space(3))) v4i* lptr4;   // LDS 128-bit chunk

#define NHEADS 16
#define HDIM   64
#define DMODEL 1024
#define BB     2
#define LL     2048

#if __has_builtin(__builtin_amdgcn_global_load_async_to_lds_b128) && \
    __has_builtin(__builtin_amdgcn_s_wait_asynccnt)
#define USE_ASYNC_LDS 1
#else
#define USE_ASYNC_LDS 0
#endif

// ---------------------------------------------------------------------------
// Block-tiled GEMM: C(MxN,f32) = A(MxK,f32) * B(KxN,f32), bf16 WMMA, f32 acc.
// Block = 256 threads (8 waves) -> 128x64 C tile, BK=32 k-steps.
// Wave (4x2 grid) -> 32x32 sub-tile = 2x2 accumulators = 4 WMMA per k-step.
// Double-buffered LDS staging via GLOBAL_LOAD_ASYNC_TO_LDS_B128:
// DMA of step s+1 overlaps WMMA of step s; one s_wait_asynccnt+barrier/iter.
// ---------------------------------------------------------------------------
#define BKDEPTH 32
#define ASTRIDE 36   // 144B row stride: 16B-aligned rows, conflict-free frag reads
#define BNW     64

__global__ __launch_bounds__(256)
void gemm_bf16_wmma(const float* __restrict__ A, const float* __restrict__ Bm,
                    float* __restrict__ C, int M, int N, int K) {
    __shared__ float Asl[2][128 * ASTRIDE];   // 2 x 18KB
    __shared__ float Bsl[2][BKDEPTH * BNW];   // 2 x 8KB

    const int tid  = threadIdx.x;
    const int lane = tid & 31;
    const int wave = tid >> 5;
    const int lm   = lane & 15;
    const int half = lane >> 4;
    const int Nt   = N / BNW;
    const int bm0  = (int)(blockIdx.x / Nt) << 7;
    const int bn0  = (int)(blockIdx.x % Nt) * BNW;
    const int wm   = (wave >> 1) << 5;   // 0 / 32 / 64 / 96
    const int wn   = (wave & 1) << 5;    // 0 / 32

    const v8f vzero = {0.f,0.f,0.f,0.f,0.f,0.f,0.f,0.f};
    v8f acc[2][2];
#pragma unroll
    for (int g = 0; g < 2; ++g)
#pragma unroll
        for (int t = 0; t < 2; ++t) acc[g][t] = vzero;

    // staging: A = 4 b128 chunks/thread, B = 2 b128 chunks/thread
    const int ar = tid >> 3;          // A row 0..31 (+c*32)
    const int ac = (tid & 7) << 2;    // A col 0..28
    const int bk = tid >> 4;          // B row 0..15 (+c*16)
    const int bn = (tid & 15) << 2;   // B col 0..60

    auto stage = [&](int k0, int buf) {
#if USE_ASYNC_LDS
#pragma unroll
        for (int c = 0; c < 4; ++c) {
            int r = ar + c * 32;
            __builtin_amdgcn_global_load_async_to_lds_b128(
                (gptr4)(A + (size_t)(bm0 + r) * K + k0 + ac),
                (lptr4)(&Asl[buf][r * ASTRIDE + ac]), 0, 0);
        }
#pragma unroll
        for (int c = 0; c < 2; ++c) {
            int kk = bk + c * 16;
            __builtin_amdgcn_global_load_async_to_lds_b128(
                (gptr4)(Bm + (size_t)(k0 + kk) * N + bn0 + bn),
                (lptr4)(&Bsl[buf][kk * BNW + bn]), 0, 0);
        }
#else
#pragma unroll
        for (int c = 0; c < 4; ++c) {
            int r = ar + c * 32;
            *(float4*)(&Asl[buf][r * ASTRIDE + ac]) =
                *(const float4*)(A + (size_t)(bm0 + r) * K + k0 + ac);
        }
#pragma unroll
        for (int c = 0; c < 2; ++c) {
            int kk = bk + c * 16;
            *(float4*)(&Bsl[buf][kk * BNW + bn]) =
                *(const float4*)(Bm + (size_t)(k0 + kk) * N + bn0 + bn);
        }
#endif
    };

    const int nsteps = K / BKDEPTH;
    stage(0, 0);

    for (int s = 0; s < nsteps; ++s) {
        const int cur = s & 1;
#if USE_ASYNC_LDS
        __builtin_amdgcn_s_wait_asynccnt(0);
#endif
        __syncthreads();   // buf[cur] visible to all; all done reading buf[cur^1]
        if (s + 1 < nsteps) stage((s + 1) * BKDEPTH, cur ^ 1);  // overlap with compute

        // A fragments: 2 row-groups of 16 rows, K=32 bf16
        v16bf afr[2];
#pragma unroll
        for (int g = 0; g < 2; ++g) {
            const int row = wm + g * 16 + lm;
#pragma unroll
            for (int v = 0; v < 8; ++v) {
                int ka = ((v >> 2) << 4) + (half << 3) + ((v & 3) << 1);
                v2f av = *(const v2f*)(&Asl[cur][row * ASTRIDE + ka]);
                afr[g][2*v]   = (__bf16)av.x;
                afr[g][2*v+1] = (__bf16)av.y;
            }
        }
        // B fragments: 2 column-groups of 16 cols
        v16bf bfr[2];
#pragma unroll
        for (int t = 0; t < 2; ++t) {
            const int n = wn + t * 16 + lm;
#pragma unroll
            for (int v = 0; v < 8; ++v) {
                int kb = ((v >> 2) << 4) + (half << 3) + ((v & 3) << 1);
                bfr[t][2*v]   = (__bf16)Bsl[cur][kb * BNW + n];
                bfr[t][2*v+1] = (__bf16)Bsl[cur][(kb + 1) * BNW + n];
            }
        }
#pragma unroll
        for (int g = 0; g < 2; ++g)
#pragma unroll
            for (int t = 0; t < 2; ++t)
                acc[g][t] = __builtin_amdgcn_wmma_f32_16x16x32_bf16(
                    false, afr[g], false, bfr[t], (short)0, acc[g][t], false, false);
    }

#pragma unroll
    for (int g = 0; g < 2; ++g)
#pragma unroll
        for (int t = 0; t < 2; ++t)
#pragma unroll
            for (int r = 0; r < 8; ++r)
                C[(size_t)(bm0 + wm + g*16 + r + 8*half) * N + bn0 + wn + t*16 + lm]
                    = acc[g][t][r];
}

// ---------------------------------------------------------------------------
// Per-head LayerNorm + RoPE on q,k; scatter q,k,v into [B,H,L,64] layout.
// ---------------------------------------------------------------------------
__global__ __launch_bounds__(256)
void normrope_kernel(const float* __restrict__ qkv, const int* __restrict__ pos,
                     const float* __restrict__ qn_w, const float* __restrict__ kn_w,
                     float* __restrict__ Qp, float* __restrict__ Kp,
                     float* __restrict__ Vp) {
    int tid = blockIdx.x * blockDim.x + threadIdx.x;
    if (tid >= BB * NHEADS * LL) return;
    int l = tid % LL;
    int h = (tid / LL) % NHEADS;
    int b = tid / (NHEADS * LL);
    const size_t rowbase = (size_t)(b * LL + l) * 3 * DMODEL;
    const size_t obase   = ((size_t)(b * NHEADS + h) * LL + l) * HDIM;
    const float  p       = (float)pos[b * LL + l];

    for (int which = 0; which < 2; ++which) {
        const float* src = qkv + rowbase + which * DMODEL + h * HDIM;
        const float* w   = (which == 0 ? qn_w : kn_w) + h * HDIM;
        float*       dst = (which == 0 ? Qp : Kp) + obase;
        float x[HDIM];
        float sum = 0.f;
#pragma unroll
        for (int d = 0; d < HDIM; ++d) { x[d] = src[d]; sum += x[d]; }
        float mu = sum * (1.f / HDIM);
        float var = 0.f;
#pragma unroll
        for (int d = 0; d < HDIM; ++d) { float t = x[d] - mu; var += t * t; }
        var *= (1.f / HDIM);
        float rs = rsqrtf(var + 1e-6f);
#pragma unroll
        for (int d = 0; d < HDIM; ++d) x[d] = (x[d] - mu) * rs * w[d];
#pragma unroll
        for (int i = 0; i < HDIM / 2; ++i) {
            float inv_freq = expf(-((2.f * i) / (float)HDIM) * logf(10000.f));
            float ang = p * inv_freq;
            float c = cosf(ang), s = sinf(ang);
            float e0 = x[2*i], e1 = x[2*i+1];
            dst[2*i]   = e0 * c - e1 * s;
            dst[2*i+1] = e1 * c + e0 * s;
        }
    }
    const float* vsrc = qkv + rowbase + 2 * DMODEL + h * HDIM;
    float*       vdst = Vp + obase;
#pragma unroll
    for (int d = 0; d < HDIM; ++d) vdst[d] = vsrc[d];
}

// ---------------------------------------------------------------------------
// Flash attention: one wave handles 16 query rows of one (b,h).
// QK^T in bf16 WMMA (f32 acc), online softmax in registers,
// P*V in full-precision v_wmma_f32_16x16x4_f32 (P transposed via per-wave LDS).
// ---------------------------------------------------------------------------
__global__ __launch_bounds__(256)
void flash_attn_kernel(const float* __restrict__ Qp, const float* __restrict__ Kp,
                       const float* __restrict__ Vp,
                       const unsigned char* __restrict__ mask,
                       float* __restrict__ Aout) {
    __shared__ float lds[8][16 * 17];
    const int lane = threadIdx.x & 31;
    const int wave = threadIdx.x >> 5;
    const int lm   = lane & 15;
    const int half = lane >> 4;
    const int LB = LL / 16;
    int gid = blockIdx.x * 8 + wave;
    if (gid >= BB * NHEADS * LB) return;
    int mblk = gid % LB;
    int h    = (gid / LB) % NHEADS;
    int b    = gid / (NHEADS * LB);
    const size_t bh = (size_t)(b * NHEADS + h) * LL;
    const int m0 = mblk * 16;
    float* myLds = &lds[wave][0];

    v16bf qa[2];
#pragma unroll
    for (int f = 0; f < 2; ++f) {
        const float* qrow = Qp + (bh + m0 + lm) * HDIM + f * 32;
#pragma unroll
        for (int v = 0; v < 8; ++v) {
            int ka = ((v >> 2) << 4) + (half << 3) + ((v & 3) << 1);
            v2f av = *(const v2f*)(qrow + ka);
            qa[f][2*v]   = (__bf16)av.x;
            qa[f][2*v+1] = (__bf16)av.y;
        }
    }

    const v8f vzero = {0.f,0.f,0.f,0.f,0.f,0.f,0.f,0.f};
    v8f o[4];
#pragma unroll
    for (int t = 0; t < 4; ++t) o[t] = vzero;
    float m8[8], l8[8];
#pragma unroll
    for (int r = 0; r < 8; ++r) { m8[r] = -INFINITY; l8[r] = 0.f; }
    const float scale = 0.125f;

    for (int j = 0; j < LB; ++j) {
        v16bf kb0, kb1;
        const float* krow = Kp + (bh + j * 16 + lm) * HDIM;
#pragma unroll
        for (int v = 0; v < 8; ++v) {
            int kk = ((v >> 2) << 4) + (half << 3) + ((v & 3) << 1);
            v2f k0v = *(const v2f*)(krow + kk);
            kb0[2*v] = (__bf16)k0v.x; kb0[2*v+1] = (__bf16)k0v.y;
            v2f k1v = *(const v2f*)(krow + 32 + kk);
            kb1[2*v] = (__bf16)k1v.x; kb1[2*v+1] = (__bf16)k1v.y;
        }
        v8f s = vzero;
        s = __builtin_amdgcn_wmma_f32_16x16x32_bf16(false, qa[0], false, kb0,
                                                    (short)0, s, false, false);
        s = __builtin_amdgcn_wmma_f32_16x16x32_bf16(false, qa[1], false, kb1,
                                                    (short)0, s, false, false);
        float sm[8];
#pragma unroll
        for (int r = 0; r < 8; ++r) {
            int row = m0 + r + 8 * half;
            unsigned char mk = mask[((size_t)b * LL + row) * LL + j * 16 + lm];
            sm[r] = mk ? s[r] * scale : -INFINITY;
        }
        float pr[8], corr[8];
#pragma unroll
        for (int r = 0; r < 8; ++r) {
            float mx = sm[r];
            mx = fmaxf(mx, __shfl_xor(mx, 1, 32));
            mx = fmaxf(mx, __shfl_xor(mx, 2, 32));
            mx = fmaxf(mx, __shfl_xor(mx, 4, 32));
            mx = fmaxf(mx, __shfl_xor(mx, 8, 32));
            float mnew = fmaxf(m8[r], mx);
            float p, c;
            if (mnew == -INFINITY) { p = 0.f; c = 1.f; }
            else { p = __expf(sm[r] - mnew); c = __expf(m8[r] - mnew); }
            float rs = p;
            rs += __shfl_xor(rs, 1, 32);
            rs += __shfl_xor(rs, 2, 32);
            rs += __shfl_xor(rs, 4, 32);
            rs += __shfl_xor(rs, 8, 32);
            l8[r] = l8[r] * c + rs;
            m8[r] = mnew;
            pr[r] = p;
            corr[r] = c;
        }
#pragma unroll
        for (int t = 0; t < 4; ++t)
#pragma unroll
            for (int r = 0; r < 8; ++r) o[t][r] *= corr[r];
#pragma unroll
        for (int r = 0; r < 8; ++r)
            myLds[(r + 8 * half) * 17 + lm] = pr[r];
#pragma unroll
        for (int kbk = 0; kbk < 4; ++kbk) {
            v2f pa;
            pa.x = myLds[lm * 17 + kbk * 4 + 2 * half + 0];
            pa.y = myLds[lm * 17 + kbk * 4 + 2 * half + 1];
            const float* vbase = Vp + (bh + j * 16 + kbk * 4 + 2 * half) * HDIM;
#pragma unroll
            for (int t = 0; t < 4; ++t) {
                v2f vb;
                vb.x = vbase[t * 16 + lm];
                vb.y = vbase[HDIM + t * 16 + lm];
                o[t] = __builtin_amdgcn_wmma_f32_16x16x4_f32(
                           false, pa, false, vb, (short)0, o[t], false, false);
            }
        }
    }
#pragma unroll
    for (int r = 0; r < 8; ++r) {
        float inv = 1.f / l8[r];
        int row = m0 + r + 8 * half;
        float* orow = Aout + ((size_t)(b * LL) + row) * DMODEL + h * HDIM;
#pragma unroll
        for (int t = 0; t < 4; ++t)
            orow[t * 16 + lm] = o[t][r] * inv;
    }
}

// ---------------------------------------------------------------------------
extern "C" void kernel_launch(void* const* d_in, const int* in_sizes, int n_in,
                              void* d_out, int out_size, void* d_ws, size_t ws_size,
                              hipStream_t stream) {
    const float*         x     = (const float*)d_in[0];
    const unsigned char* mask  = (const unsigned char*)d_in[1];
    const int*           posi  = (const int*)d_in[2];
    const float*         W_qkv = (const float*)d_in[3];
    const float*         W_out = (const float*)d_in[4];
    const float*         qn_w  = (const float*)d_in[5];
    const float*         kn_w  = (const float*)d_in[6];
    float*               out   = (float*)d_out;

    float* ws  = (float*)d_ws;
    const size_t rows = (size_t)BB * LL;              // 4096
    float* qkv = ws;                                  // rows * 3072 f32
    float* Qp  = ws + rows * 3 * DMODEL;
    float* Kp  = Qp + (size_t)BB * NHEADS * LL * HDIM;
    float* Vp  = Kp + (size_t)BB * NHEADS * LL * HDIM;
    float* attn_out = qkv;  // reuse: qkv is dead after normrope

    // 1) qkv = x @ W_qkv  (4096 x 3072, K=1024): 32x48 = 1536 blocks
    {
        int M = (int)rows, N = 3 * DMODEL, K = DMODEL;
        int blocks = (M / 128) * (N / BNW);
        gemm_bf16_wmma<<<blocks, 256, 0, stream>>>(x, W_qkv, qkv, M, N, K);
    }
    // 2) per-head LN + RoPE, scatter to [B,H,L,64]
    {
        int total = BB * NHEADS * LL;
        normrope_kernel<<<(total + 255) / 256, 256, 0, stream>>>(
            qkv, posi, qn_w, kn_w, Qp, Kp, Vp);
    }
    // 3) flash attention -> attn_out [B,L,1024]
    {
        int waves = BB * NHEADS * (LL / 16);
        flash_attn_kernel<<<(waves + 7) / 8, 256, 0, stream>>>(
            Qp, Kp, Vp, mask, attn_out);
    }
    // 4) out = attn_out @ W_out  (4096 x 1024, K=1024): 32x16 = 512 blocks
    {
        int M = (int)rows, N = DMODEL, K = DMODEL;
        int blocks = (M / 128) * (N / BNW);
        gemm_bf16_wmma<<<blocks, 256, 0, stream>>>(attn_out, W_out, out, M, N, K);
    }
}
